// PreactivResBlock_bn_hyper_TTT_53369263620502
// MI455X (gfx1250) — matrix operand order repaired and checked by reading, not compiled, not run on hardware
//
#include <hip/hip_runtime.h>
#include <hip/hip_bf16.h>
#include <stdint.h>

// ---------------------------------------------------------------------------
// Types / helpers
// ---------------------------------------------------------------------------
typedef unsigned short u16b;
typedef __attribute__((ext_vector_type(16))) __bf16 v16bf;
typedef __attribute__((ext_vector_type(8)))  float  v8f;

union Frag16 { uint4 q[2]; v16bf v; };

__device__ __forceinline__ u16b f2bf(float f) {
  union { float f; uint32_t u; } c; c.f = f;
  return (u16b)((c.u + 0x7fffu + ((c.u >> 16) & 1u)) >> 16);   // RNE
}
__device__ __forceinline__ float bf2f(u16b h) {
  union { uint32_t u; float f; } c; c.u = ((uint32_t)h) << 16;
  return c.f;
}

// Raw LDS byte offset of a __shared__ pointer (AS3 pointers are 32-bit offsets)
__device__ __forceinline__ unsigned lds_addr(const void* p) {
  return (unsigned)(unsigned long long)(__attribute__((address_space(3))) const void*)p;
}

// 16-byte async copy global -> LDS (ASYNCcnt tracked, bypasses VGPRs)
__device__ __forceinline__ void async_copy_b128(unsigned ldsOff, const void* gptr) {
  asm volatile("global_load_async_to_lds_b128 %0, %1, off"
               :: "v"(ldsOff), "v"((unsigned long long)(uintptr_t)gptr)
               : "memory");
}
__device__ __forceinline__ void async_wait0() {
  asm volatile("s_wait_asynccnt 0" ::: "memory");
}

// ---------------------------------------------------------------------------
// Hypernetwork: params[p] = genw[p,:] . prelu(fc1(prelu(fc0(feat)))) + genb[p]
// Weights written as bf16 GEMM-A layout:  wA[b][co][K], K = tap*cin + ci,
// K padded to kpad (pad pre-zeroed by memset).  Biases kept fp32.
// ---------------------------------------------------------------------------
__global__ void hyper_gen(const float* __restrict__ feat,
                          const float* __restrict__ fc0w, const float* __restrict__ fc0b,
                          const float* __restrict__ a0,
                          const float* __restrict__ fc1w, const float* __restrict__ fc1b,
                          const float* __restrict__ a1,
                          const float* __restrict__ genw, const float* __restrict__ genb,
                          int cin, int taps, int kpad,
                          u16b* __restrict__ wA, float* __restrict__ bias)
{
  const int COUT = 32;
  int b = blockIdx.x;
  const float* f = feat + b * 10;
  float h0[16];
#pragma unroll
  for (int i = 0; i < 16; ++i) {
    float s = fc0b[i];
#pragma unroll
    for (int j = 0; j < 10; ++j) s += f[j] * fc0w[i * 10 + j];
    h0[i] = s >= 0.f ? s : a0[0] * s;
  }
  float h1v[8];
#pragma unroll
  for (int i = 0; i < 8; ++i) {
    float s = fc1b[i];
#pragma unroll
    for (int j = 0; j < 16; ++j) s += h0[j] * fc1w[i * 16 + j];
    h1v[i] = s >= 0.f ? s : a1[0] * s;
  }
  int nw = COUT * cin * taps;
  int nparam = nw + COUT;
  for (int p = threadIdx.x; p < nparam; p += blockDim.x) {
    float s = genb[p];
#pragma unroll
    for (int q = 0; q < 8; ++q) s += genw[p * 8 + q] * h1v[q];
    if (p < nw) {
      int co = p / (cin * taps);
      int rem = p - co * cin * taps;
      int ci = rem / taps;
      int tap = rem - ci * taps;
      wA[(size_t)b * COUT * kpad + (size_t)co * kpad + tap * cin + ci] = f2bf(s);
    } else {
      bias[b * COUT + (p - nw)] = s;
    }
  }
}

// ---------------------------------------------------------------------------
// Per-channel sum / sumsq over data[b][c][s]  (for BN batch stats)
// ---------------------------------------------------------------------------
__global__ void chan_reduce(const float* __restrict__ data, int C, int S, int B,
                            int perSlice, float* __restrict__ sum, float* __restrict__ sq)
{
  int c = blockIdx.x;
  long long total = (long long)B * S;
  long long start = (long long)blockIdx.y * perSlice;
  long long end = start + perSlice; if (end > total) end = total;
  float s0 = 0.f, s1 = 0.f;
  for (long long t = start + threadIdx.x; t < end; t += blockDim.x) {
    long long b = t / S;
    long long s = t - b * S;
    float v = data[(b * C + c) * (long long)S + s];
    s0 += v; s1 += v * v;
  }
  __shared__ float r0[256], r1[256];
  r0[threadIdx.x] = s0; r1[threadIdx.x] = s1;
  __syncthreads();
  for (int o = 128; o > 0; o >>= 1) {
    if ((int)threadIdx.x < o) { r0[threadIdx.x] += r0[threadIdx.x + o]; r1[threadIdx.x] += r1[threadIdx.x + o]; }
    __syncthreads();
  }
  if (threadIdx.x == 0) { atomicAdd(&sum[c], r0[0]); atomicAdd(&sq[c], r1[0]); }
}

__global__ void bn_finalize(const float* __restrict__ sum, const float* __restrict__ sq,
                            const float* __restrict__ g, const float* __restrict__ bb,
                            int C, float invCount,
                            float* __restrict__ scale, float* __restrict__ shift)
{
  int c = threadIdx.x;
  if (c < C) {
    float m = sum[c] * invCount;
    float v = sq[c] * invCount - m * m;
    float sc = g[c] * rsqrtf(v + 1e-5f);
    scale[c] = sc;
    shift[c] = bb[c] - m * sc;
  }
}

// ---------------------------------------------------------------------------
// NCDHW fp32 -> NDHWC bf16 with fused bn+relu.  One block per (b, row) where a
// row is one (z,y) line: C channels x W x-positions (C*W == 1024 for both uses).
// Reads coalesced from NCDHW, transpose in LDS, contiguous 16B writes NDHWC.
// ---------------------------------------------------------------------------
__global__ __launch_bounds__(256) void bnrelu_ndhwc(const float* __restrict__ in,
                                                    const float* __restrict__ scale,
                                                    const float* __restrict__ shift,
                                                    int C, int wLog2, int rLog2,
                                                    u16b* __restrict__ out)
{
  __shared__ __align__(16) u16b row[1024];
  int W = 1 << wLog2;
  int nRows = 1 << rLog2;
  int blk = blockIdx.x;
  int b = blk >> rLog2;
  int r = blk & (nRows - 1);
  int CW = C * W;                                    // == 1024
  for (int idx = threadIdx.x; idx < CW; idx += 256) {
    int c = idx >> wLog2, xq = idx & (W - 1);
    float v = in[((size_t)(b * C + c) << rLog2 | r) * W + xq];
    v = v * scale[c] + shift[c];
    row[xq * C + c] = f2bf(v > 0.f ? v : 0.f);
  }
  __syncthreads();
  uint4* dst = (uint4*)(out + ((size_t)(b << rLog2 | r)) * CW);
  const uint4* srcv = (const uint4*)row;
  for (int idx = threadIdx.x; idx < (CW >> 3); idx += 256)
    dst[idx] = srcv[idx];
}

// ---------------------------------------------------------------------------
// identity = conv1x1 stride2 on RAW x (fp32 MACs; only 0.27 GFLOP, BW bound)
// ---------------------------------------------------------------------------
__global__ __launch_bounds__(256) void convd_1x1(const float* __restrict__ x,
                                                 const u16b* __restrict__ wAd,
                                                 const float* __restrict__ bd,
                                                 float* __restrict__ iden)
{
  int b = blockIdx.z, co = blockIdx.y;
  __shared__ float w[16];
  if (threadIdx.x < 16) w[threadIdx.x] = bf2f(wAd[(b * 32 + co) * 16 + threadIdx.x]);
  __syncthreads();
  int n = blockIdx.x * 256 + threadIdx.x;            // 0..32767
  int zo = n >> 10, yo = (n >> 5) & 31, xo = n & 31;
  size_t base = (size_t)(b * 16) * 262144 + (size_t)(2 * zo) * 4096 + (2 * yo) * 64 + 2 * xo;
  float s = bd[b * 32 + co];
#pragma unroll
  for (int ci = 0; ci < 16; ++ci) s += x[base + (size_t)ci * 262144] * w[ci];
  iden[(size_t)(b * 32 + co) * 32768 + n] = s;
}

// ---------------------------------------------------------------------------
// conv1: 3x3x3 stride2 pad1, 16 -> 32, input xb in NDHWC bf16 (bn1+relu).
// Implicit GEMM:  M=32(co), N=32768(spatial), K=432 (tap*16+ci), Kpad=448.
// Block tile: one z-plane, 4 y-rows x 32 x (N=128), full M.  8 waves = 2x4,
// each wave: 16(M) x 32(N) via two f32 16x16 accumulators, 14 K-chunks of 32
// (tap pair 2j / 2j+1; fictitious tap 27 -> zeroed LDS pad, EXEC stays full).
// LDS patch [z=3][y=10][x=66][c=16] == NDHWC slices -> pure async b128 copies.
// K-loop fully unrolled so tap decomposition folds to constant LDS offsets.
// ---------------------------------------------------------------------------
#define C1_POS  (3 * 10 * 66)
#define C1_LDSN (C1_POS * 16)   // 31680 halves

__global__ __launch_bounds__(256) void conv1_wmma(const u16b* __restrict__ xb,
                                                  const u16b* __restrict__ wA,
                                                  const float* __restrict__ b1,
                                                  float* __restrict__ out1)
{
  __shared__ __align__(16) u16b smem[C1_LDSN + 16];
  int b = blockIdx.y;
  int zo = blockIdx.x >> 3, yt = blockIdx.x & 7;
  int tid = threadIdx.x;
  int zi0 = 2 * zo - 1, yi0 = 8 * yt - 1;
  const u16b* src = xb + (size_t)b * 64 * 64 * 64 * 16;

  // 30 (z,y) rows x 132 16B-chunks (66 x-positions, 2 chunks of 8 ch each)
  for (int idx = tid; idx < 30 * 132; idx += 256) {
    int rrow = idx / 132, ch = idx - rrow * 132;
    int y = rrow % 10, z = rrow / 10;
    int px = ch >> 1, half = ch & 1;
    int gx = px - 1, gy = yi0 + y, gz = zi0 + z;
    unsigned elem = (unsigned)((((z * 10 + y) * 66 + px) << 4) + half * 8);
    if ((unsigned)gx < 64u && (unsigned)gy < 64u && (unsigned)gz < 64u) {
      const u16b* gp = src + (((((size_t)gz * 64 + gy) * 64 + gx) << 4) + half * 8);
      async_copy_b128(lds_addr(&smem[elem]), gp);
    } else {
      *(uint4*)&smem[elem] = make_uint4(0, 0, 0, 0);
    }
  }
  if (tid < 16) smem[C1_LDSN + tid] = 0;             // zero pad fragment (tap 27)
  async_wait0();
  __syncthreads();

  int wv = tid >> 5, lane = tid & 31;
  int mi = wv >> 2, ni = wv & 3;                      // 2 (M) x 4 (N=y-row) waves
  int lcol = lane & 15, lhalf = lane >> 4;
  v8f acc0 = {0,0,0,0,0,0,0,0}, acc1 = acc0;
  const u16b* wbase = wA + (size_t)b * 32 * 448;
  int Mrow = mi * 16 + lcol;                          // A: M = lane%16

#pragma unroll
  for (int j = 0; j < 14; ++j) {
    Frag16 Af, Bf0, Bf1;
    // A frag: lanes<16 K{0-15}, lanes>=16 K{16-31}  -> lhalf*16 offset
    const u16b* ap = wbase + (size_t)Mrow * 448 + j * 32 + lhalf * 16;
    Af.q[0] = *(const uint4*)ap;
    Af.q[1] = *(const uint4*)(ap + 8);
    // B frag: lanes<16 tap 2j, lanes>=16 tap 2j+1; 16 contiguous channels
    int tap = 2 * j + lhalf;
    if (tap < 27) {
      int tz = tap / 9, tr = tap % 9, ty = tr / 3, tx = tr % 3;
      int yrow = 2 * ni + ty;
      const u16b* p0 = &smem[((tz * 10 + yrow) * 66 + 2 * lcol + tx) * 16];
      const u16b* p1 = &smem[((tz * 10 + yrow) * 66 + 2 * (16 + lcol) + tx) * 16];
      Bf0.q[0] = *(const uint4*)p0; Bf0.q[1] = *(const uint4*)(p0 + 8);
      Bf1.q[0] = *(const uint4*)p1; Bf1.q[1] = *(const uint4*)(p1 + 8);
    } else {
      const u16b* pz = &smem[C1_LDSN];
      Bf0.q[0] = *(const uint4*)pz; Bf0.q[1] = *(const uint4*)(pz + 8);
      Bf1 = Bf0;
    }
    acc0 = __builtin_amdgcn_wmma_f32_16x16x32_bf16(false, Af.v, false, Bf0.v, (short)0, acc0, false, false);
    acc1 = __builtin_amdgcn_wmma_f32_16x16x32_bf16(false, Af.v, false, Bf1.v, (short)0, acc1, false, false);
  }

  // C/D layout: VGPR r -> M = r + 8*lhalf ; N = lane%16
  size_t outBase = (size_t)(b * 32) * 32768 + (size_t)zo * 1024 + (yt * 4 + ni) * 32;
#pragma unroll
  for (int r = 0; r < 8; ++r) {
    int co = mi * 16 + lhalf * 8 + r;
    float bv = b1[b * 32 + co];
    out1[outBase + (size_t)co * 32768 + lcol]      = acc0[r] + bv;
    out1[outBase + (size_t)co * 32768 + 16 + lcol] = acc1[r] + bv;
  }
}

// ---------------------------------------------------------------------------
// conv2: 3x3x3 stride1 pad1, 32 -> 32, input out1b NDHWC bf16 (bn2+relu).
// K = 27 taps * 32 ch = 864 -> 27 exact chunks of 32, fully unrolled so all
// LDS B-fragment addresses become base + constant offset.
// Epilogue fuses + b2 + BN_d(identity) and writes d_out (fp32 NCDHW).
// LDS patch [z=3][y=6][x=34][c=32].
// ---------------------------------------------------------------------------
#define C2_POS  (3 * 6 * 34)
#define C2_LDSN (C2_POS * 32)   // 19584 halves

__global__ __launch_bounds__(256) void conv2_wmma(const u16b* __restrict__ inb,
                                                  const u16b* __restrict__ wA,
                                                  const float* __restrict__ b2,
                                                  const float* __restrict__ iden,
                                                  const float* __restrict__ scd,
                                                  const float* __restrict__ shd,
                                                  float* __restrict__ out)
{
  __shared__ __align__(16) u16b smem[C2_LDSN];
  int b = blockIdx.y;
  int zo = blockIdx.x >> 3, yt = blockIdx.x & 7;
  int tid = threadIdx.x;
  int zi0 = zo - 1, yi0 = 4 * yt - 1;
  const u16b* src = inb + (size_t)b * 32 * 32768;

  // 18 (z,y) rows x 136 16B-chunks (34 x-positions, 4 chunks of 8 ch each)
  for (int idx = tid; idx < 18 * 136; idx += 256) {
    int rrow = idx / 136, ch = idx - rrow * 136;
    int y = rrow % 6, z = rrow / 6;
    int px = ch >> 2, q = ch & 3;
    int gx = px - 1, gy = yi0 + y, gz = zi0 + z;
    unsigned elem = (unsigned)((((z * 6 + y) * 34 + px) << 5) + q * 8);
    if ((unsigned)gx < 32u && (unsigned)gy < 32u && (unsigned)gz < 32u) {
      const u16b* gp = src + (((((size_t)gz * 32 + gy) * 32 + gx) << 5) + q * 8);
      async_copy_b128(lds_addr(&smem[elem]), gp);
    } else {
      *(uint4*)&smem[elem] = make_uint4(0, 0, 0, 0);
    }
  }
  async_wait0();
  __syncthreads();

  int wv = tid >> 5, lane = tid & 31;
  int mi = wv >> 2, ni = wv & 3;
  int lcol = lane & 15, lhalf = lane >> 4;
  v8f acc0 = {0,0,0,0,0,0,0,0}, acc1 = acc0;
  const u16b* wbase = wA + (size_t)b * 32 * 864;
  int Mrow = mi * 16 + lcol;

#pragma unroll
  for (int j = 0; j < 27; ++j) {
    Frag16 Af, Bf0, Bf1;
    const u16b* ap = wbase + (size_t)Mrow * 864 + j * 32 + lhalf * 16;
    Af.q[0] = *(const uint4*)ap;
    Af.q[1] = *(const uint4*)(ap + 8);
    int tz = j / 9, tr = j % 9, ty = tr / 3, tx = tr % 3;
    int yrow = ni + ty;
    // chunk K = 32 channels of tap j; lanes<16 -> ch 0-15, lanes>=16 -> ch 16-31
    const u16b* p0 = &smem[((tz * 6 + yrow) * 34 + lcol + tx) * 32 + lhalf * 16];
    const u16b* p1 = &smem[((tz * 6 + yrow) * 34 + 16 + lcol + tx) * 32 + lhalf * 16];
    Bf0.q[0] = *(const uint4*)p0; Bf0.q[1] = *(const uint4*)(p0 + 8);
    Bf1.q[0] = *(const uint4*)p1; Bf1.q[1] = *(const uint4*)(p1 + 8);
    acc0 = __builtin_amdgcn_wmma_f32_16x16x32_bf16(false, Af.v, false, Bf0.v, (short)0, acc0, false, false);
    acc1 = __builtin_amdgcn_wmma_f32_16x16x32_bf16(false, Af.v, false, Bf1.v, (short)0, acc1, false, false);
  }

  size_t outBase = (size_t)(b * 32) * 32768 + (size_t)zo * 1024 + (yt * 4 + ni) * 32;
#pragma unroll
  for (int r = 0; r < 8; ++r) {
    int co = mi * 16 + lhalf * 8 + r;
    float bv = b2[b * 32 + co];
    float sc = scd[co], sh = shd[co];
    size_t i0 = outBase + (size_t)co * 32768 + lcol;
    size_t i1 = i0 + 16;
    out[i0] = acc0[r] + bv + iden[i0] * sc + sh;
    out[i1] = acc1[r] + bv + iden[i1] * sc + sh;
  }
}

// ---------------------------------------------------------------------------
// Launch
// ---------------------------------------------------------------------------
extern "C" void kernel_launch(void* const* d_in, const int* in_sizes, int n_in,
                              void* d_out, int out_size, void* d_ws, size_t ws_size,
                              hipStream_t stream)
{
  (void)in_sizes; (void)n_in; (void)out_size; (void)ws_size;
  const float* x     = (const float*)d_in[0];
  const float* feat  = (const float*)d_in[1];
  const float* bn1_g = (const float*)d_in[2];
  const float* bn1_b = (const float*)d_in[3];
  const float* bn2_g = (const float*)d_in[4];
  const float* bn2_b = (const float*)d_in[5];
  const float* bnd_g = (const float*)d_in[6];
  const float* bnd_b = (const float*)d_in[7];
#define HP(base, k) ((const float*)d_in[(base) + (k)])

  char* ws = (char*)d_ws;
  size_t off = 0;
  auto take = [&](size_t bytes) -> void* {
    void* p = ws + off;
    off += (bytes + 255) & ~(size_t)255;
    return p;
  };
  u16b*  xb    = (u16b*) take(67108864ull);           // bn1+relu(x), bf16 NDHWC
  float* out1  = (float*)take(33554432ull);           // conv1 out, fp32 NCDHW
  u16b*  out1b = (u16b*) take(16777216ull);           // bn2+relu(out1), bf16 NDHWC
  float* iden  = (float*)take(33554432ull);           // identity, fp32 NCDHW
  u16b*  wA1   = (u16b*) take(8ull * 32 * 448 * 2);
  u16b*  wA2   = (u16b*) take(8ull * 32 * 864 * 2);
  u16b*  wAd   = (u16b*) take(8ull * 32 * 16  * 2);
  float* b1    = (float*)take(8 * 32 * 4);
  float* b2    = (float*)take(8 * 32 * 4);
  float* bd    = (float*)take(8 * 32 * 4);
  float* stats = (float*)take(160 * 4);
  float* scsh  = (float*)take(160 * 4);
  float *sum1 = stats,      *sq1 = stats + 16;
  float *sum2 = stats + 32, *sq2 = stats + 64;
  float *sumd = stats + 96, *sqd = stats + 128;
  float *sc1 = scsh,      *sh1 = scsh + 16;
  float *sc2 = scsh + 32, *sh2 = scsh + 64;
  float *scd = scsh + 96, *shd = scsh + 128;

  hipMemsetAsync(wA1, 0, 8ull * 32 * 448 * 2, stream);   // zero K pad 432..447
  hipMemsetAsync(stats, 0, 160 * 4, stream);

  // Hypernet weight generation (per-sample weights + biases)
  hyper_gen<<<8, 256, 0, stream>>>(feat, HP(8,0), HP(8,1), HP(8,2), HP(8,3), HP(8,4), HP(8,5), HP(8,6), HP(8,7),
                                   16, 27, 448, wA1, b1);
  hyper_gen<<<8, 256, 0, stream>>>(feat, HP(16,0), HP(16,1), HP(16,2), HP(16,3), HP(16,4), HP(16,5), HP(16,6), HP(16,7),
                                   32, 27, 864, wA2, b2);
  hyper_gen<<<8, 256, 0, stream>>>(feat, HP(24,0), HP(24,1), HP(24,2), HP(24,3), HP(24,4), HP(24,5), HP(24,6), HP(24,7),
                                   16, 1, 16, wAd, bd);

  // BN1 stats on x, then fused bn+relu+bf16, transposed to NDHWC
  chan_reduce<<<dim3(16, 64), 256, 0, stream>>>(x, 16, 262144, 8, 32768, sum1, sq1);
  bn_finalize<<<1, 64, 0, stream>>>(sum1, sq1, bn1_g, bn1_b, 16, 1.0f / 2097152.0f, sc1, sh1);
  bnrelu_ndhwc<<<32768, 256, 0, stream>>>(x, sc1, sh1, 16, /*wLog2*/6, /*rLog2*/12, xb);

  // identity path (raw x) + its BN stats
  convd_1x1<<<dim3(128, 32, 8), 256, 0, stream>>>(x, wAd, bd, iden);
  chan_reduce<<<dim3(32, 32), 256, 0, stream>>>(iden, 32, 32768, 8, 8192, sumd, sqd);
  bn_finalize<<<1, 64, 0, stream>>>(sumd, sqd, bnd_g, bnd_b, 32, 1.0f / 262144.0f, scd, shd);

  // conv1 (WMMA), BN2 stats, bn+relu+bf16 -> NDHWC
  conv1_wmma<<<dim3(256, 8), 256, 0, stream>>>(xb, wA1, b1, out1);
  chan_reduce<<<dim3(32, 32), 256, 0, stream>>>(out1, 32, 32768, 8, 8192, sum2, sq2);
  bn_finalize<<<1, 64, 0, stream>>>(sum2, sq2, bn2_g, bn2_b, 32, 1.0f / 262144.0f, sc2, sh2);
  bnrelu_ndhwc<<<8192, 256, 0, stream>>>(out1, sc2, sh2, 32, /*wLog2*/5, /*rLog2*/10, out1b);

  // conv2 (WMMA) + fused residual-BN epilogue -> d_out
  conv2_wmma<<<dim3(256, 8), 256, 0, stream>>>(out1b, wA2, b2, iden, scd, shd, (float*)d_out);
#undef HP
}
